// TGAT_7215545057460
// MI455X (gfx1250) — matrix operand order repaired and checked by reading, not compiled, not run on hardware
//
#include <hip/hip_runtime.h>

typedef _Float16 v16h __attribute__((ext_vector_type(16)));
typedef _Float16 v8h  __attribute__((ext_vector_type(8)));
typedef float    v8f  __attribute__((ext_vector_type(8)));

#define TSTEPS 8
#define HIDDEN 128

// ---------- helpers ----------
__device__ __forceinline__ unsigned orderKey(float f) {
  unsigned u = __float_as_uint(f);
  return (u & 0x80000000u) ? ~u : (u | 0x80000000u);
}
__device__ __forceinline__ float decodeKey(unsigned k) {
  return (k & 0x80000000u) ? __uint_as_float(k ^ 0x80000000u)
                           : __uint_as_float(~k);
}
// fast sigmoid: v_exp_f32 + v_rcp_f32 (TRANS ops, co-execute with VALU)
__device__ __forceinline__ float fast_sigmoid(float z) {
  return __builtin_amdgcn_rcpf(1.0f + __expf(-z));
}
// fast tanh: (e^{2z}-1) * rcp(e^{2z}+1), clamped so inf*0 -> NaN never forms
__device__ __forceinline__ float fast_tanh(float z) {
  float z2 = fminf(fmaxf(2.0f * z, -30.0f), 30.0f);
  float t  = __expf(z2);
  return (t - 1.0f) * __builtin_amdgcn_rcpf(t + 1.0f);
}

// A fragment (16x32 f16): lane row = lane&15; halves 0..7 -> k = 8g+0..7,
// halves 8..15 -> k = 16+8g+0..7  (per ISA 16-bit A 16x32 layout)
__device__ __forceinline__ v16h ldsFragA(const _Float16* tileBase, int kbase, int lane) {
  int row = lane & 15, g = lane >> 4;
  const _Float16* p = tileBase + row * 256 + kbase + 8 * g;
  v8h lo = *(const v8h*)p;
  v8h hi = *(const v8h*)(p + 16);
  return __builtin_shufflevector(lo, hi, 0,1,2,3,4,5,6,7,8,9,10,11,12,13,14,15);
}
// B fragment (32x16 f16): lane col = lane&15; halves 0..7 -> k = 8g+0..7, halves 8..15 -> k = 16+8g+0..7.
// W is pre-transposed f16 [n][k] with k-stride 256 so each lane reads contiguous K.
__device__ __forceinline__ v16h glbFragB(const _Float16* W, int ncol0, int kbase, int lane) {
  int n = ncol0 + (lane & 15), g = lane >> 4;
  const _Float16* p = W + n * 256 + kbase + 8 * g;
  v8h lo = *(const v8h*)p;
  v8h hi = *(const v8h*)(p + 16);
  return __builtin_shufflevector(lo, hi, 0,1,2,3,4,5,6,7,8,9,10,11,12,13,14,15);
}

// ---------- one-time weight prep: f32 [k][n] -> f16 [n][k] ----------
__global__ __launch_bounds__(256) void prep_weights(const float* __restrict__ W1,
                                                    const float* __restrict__ W2,
                                                    _Float16* __restrict__ W1T,
                                                    _Float16* __restrict__ W2T) {
  int i = blockIdx.x * blockDim.x + threadIdx.x;   // 0..65535
  int n = i >> 8, k = i & 255;
  W1T[n * 256 + k] = (_Float16)W1[k * 256 + n];
  if (n < 128) W2T[n * 256 + k] = (_Float16)W2[k * 128 + n];
}

__global__ __launch_bounds__(256) void zero_h(float* __restrict__ h, int n) {
  int i = blockIdx.x * blockDim.x + threadIdx.x;
  if (i < n) h[i] = 0.0f;
}

__global__ __launch_bounds__(256) void zero_seg(unsigned* __restrict__ emaxKey,
                                                float* __restrict__ den,
                                                float* __restrict__ num, int n2) {
  int i = blockIdx.x * blockDim.x + threadIdx.x;
  if (i < n2) { emaxKey[i] = 0u; den[i] = 0.0f; num[i] = 0.0f; }
}

// ---------- GAT edge pass 1: e[edge,h] + segment max (ordered-key atomicMax) ----------
__global__ __launch_bounds__(256)
void gat_edge_pass1(const float* __restrict__ x, const int* __restrict__ ei,
                    const float* __restrict__ Wl, const float* __restrict__ Wr,
                    const float* __restrict__ bl, const float* __restrict__ br,
                    const float* __restrict__ att,
                    float* __restrict__ eScr, unsigned* __restrict__ emaxKey,
                    int nE, int nN, int t) {
  __shared__ float sWl[128], sWr[128], sB[128], sAtt[128];
  for (int i = threadIdx.x; i < 128; i += blockDim.x) {
    sWl[i] = Wl[i]; sWr[i] = Wr[i]; sB[i] = bl[i] + br[i]; sAtt[i] = att[i];
  }
  __syncthreads();
  int e = blockIdx.x * blockDim.x + threadIdx.x;
  int tot = nE + nN;
  if (e >= tot) return;
  int s, d;
  if (e < nE) { s = ei[e]; d = ei[nE + e]; } else { s = d = e - nE; }
  float a = x[s * TSTEPS + t];
  float b = x[d * TSTEPS + t];
  float acc0 = 0.0f, acc1 = 0.0f;
  #pragma unroll 8
  for (int c = 0; c < 64; ++c) {
    float v0 = fmaf(a, sWl[c], fmaf(b, sWr[c], sB[c]));
    v0 = v0 > 0.0f ? v0 : 0.2f * v0;
    acc0 = fmaf(v0, sAtt[c], acc0);
    float v1 = fmaf(a, sWl[c + 64], fmaf(b, sWr[c + 64], sB[c + 64]));
    v1 = v1 > 0.0f ? v1 : 0.2f * v1;
    acc1 = fmaf(v1, sAtt[c + 64], acc1);
  }
  eScr[2 * e]     = acc0;
  eScr[2 * e + 1] = acc1;
  atomicMax(&emaxKey[2 * d],     orderKey(acc0));
  atomicMax(&emaxKey[2 * d + 1], orderKey(acc1));
}

// ---------- GAT edge pass 2: exp + segment sums (denom, weighted num) ----------
__global__ __launch_bounds__(256)
void gat_edge_pass2(const float* __restrict__ x, const int* __restrict__ ei,
                    const float* __restrict__ eScr, const unsigned* __restrict__ emaxKey,
                    float* __restrict__ den, float* __restrict__ num,
                    int nE, int nN, int t) {
  int e = blockIdx.x * blockDim.x + threadIdx.x;
  int tot = nE + nN;
  if (e >= tot) return;
  int s, d;
  if (e < nE) { s = ei[e]; d = ei[nE + e]; } else { s = d = e - nE; }
  float a = x[s * TSTEPS + t];
  #pragma unroll
  for (int h = 0; h < 2; ++h) {
    float ev = eScr[2 * e + h];
    float m  = decodeKey(emaxKey[2 * d + h]);
    float ex = __expf(ev - m);
    unsafeAtomicAdd(&den[2 * d + h], ex);
    unsafeAtomicAdd(&num[2 * d + h], ex * a);
  }
}

// ---------- fused GAT finalize + GRU cell (WMMA) ----------
// Block: 128 threads = 4 waves; 64 rows/block. LDS: X1=[f,h] f16, X2=[f,r*h] f16.
__global__ __launch_bounds__(128)
void gru_kernel(const float* __restrict__ num, const float* __restrict__ den,
                const float* __restrict__ Wl, const float* __restrict__ bl,
                const float* __restrict__ cb, const float* __restrict__ b1,
                const float* __restrict__ b2,
                const _Float16* __restrict__ W1T, const _Float16* __restrict__ W2T,
                float* __restrict__ uBuf, float* __restrict__ h, int nN) {
  __shared__ __align__(16) _Float16 X1[64 * 256];
  __shared__ __align__(16) _Float16 X2[64 * 256];
  const int r0 = blockIdx.x * 64;

  // Phase 1: build X1 = [f | h] (f16); X2 first half = f.
  for (int i = threadIdx.x; i < 64 * 256; i += 128) {
    int row = i >> 8, col = i & 255;
    int gr = r0 + row;
    float v = 0.0f;
    if (gr < nN) {
      if (col < 128) {
        int hd = col >> 6;
        float dn = den[gr * 2 + hd];
        float inv = __builtin_amdgcn_rcpf(dn + 1e-16f);
        float S = num[gr * 2 + hd] * inv;
        float A = dn * inv;
        v = fast_sigmoid(S * Wl[col] + A * bl[col] + cb[col]);
      } else {
        v = h[gr * 128 + (col - 128)];
      }
    }
    _Float16 hv = (_Float16)v;
    X1[i] = hv;
    if (col < 128) X2[i] = hv;
  }
  __syncthreads();

  const int wave = threadIdx.x >> 5;
  const int lane = threadIdx.x & 31;
  const int g    = lane >> 4;
  const int colL = lane & 15;
  const _Float16* tileA1 = X1 + wave * 16 * 256;
  const _Float16* tileA2 = X2 + wave * 16 * 256;

  // Phase 2: ru = sigmoid(X1 @ W1 + b1); r -> X2 second half (r*h), u -> uBuf.
  for (int nt = 0; nt < 16; ++nt) {
    v8f acc = {0.0f, 0.0f, 0.0f, 0.0f, 0.0f, 0.0f, 0.0f, 0.0f};
    #pragma unroll
    for (int kt = 0; kt < 8; ++kt) {
      v16h A = ldsFragA(tileA1, kt * 32, lane);
      v16h B = glbFragB(W1T, nt * 16, kt * 32, lane);
      acc = __builtin_amdgcn_wmma_f32_16x16x32_f16(false, A, false, B,
                                                   (short)0, acc, false, false);
    }
    int col = nt * 16 + colL;
    float bias = b1[col];
    #pragma unroll
    for (int i = 0; i < 8; ++i) {
      int rl = wave * 16 + i + 8 * g;      // local row (D layout: M = i + 8g)
      int grow = r0 + rl;
      float sg = fast_sigmoid(acc[i] + bias);
      if (nt < 8) {                        // r: build r*h (f16) into X2
        float hv = (grow < nN) ? h[grow * 128 + col] : 0.0f;
        X2[rl * 256 + 128 + col] = (_Float16)(sg * hv);
      } else {                             // u: stash for phase 3
        if (grow < nN) uBuf[grow * 128 + (col - 128)] = sg;
      }
    }
  }
  __syncthreads();

  // Phase 3: c = tanh(X2 @ W2 + b2); h = u*h + (1-u)*c  (in place).
  for (int nt = 0; nt < 8; ++nt) {
    v8f acc = {0.0f, 0.0f, 0.0f, 0.0f, 0.0f, 0.0f, 0.0f, 0.0f};
    #pragma unroll
    for (int kt = 0; kt < 8; ++kt) {
      v16h A = ldsFragA(tileA2, kt * 32, lane);
      v16h B = glbFragB(W2T, nt * 16, kt * 32, lane);
      acc = __builtin_amdgcn_wmma_f32_16x16x32_f16(false, A, false, B,
                                                   (short)0, acc, false, false);
    }
    int col = nt * 16 + colL;
    float bias = b2[col];
    #pragma unroll
    for (int i = 0; i < 8; ++i) {
      int grow = r0 + wave * 16 + i + 8 * g;
      if (grow < nN) {
        float c = fast_tanh(acc[i] + bias);
        float u = uBuf[grow * 128 + col];
        float hold = h[grow * 128 + col];
        h[grow * 128 + col] = u * hold + (1.0f - u) * c;
      }
    }
  }
}

// ---------- launch ----------
extern "C" void kernel_launch(void* const* d_in, const int* in_sizes, int n_in,
                              void* d_out, int out_size, void* d_ws, size_t ws_size,
                              hipStream_t stream) {
  const float* x   = (const float*)d_in[0];
  const int*   ei  = (const int*)d_in[1];      // edge_index (2,E) int
  const float* Wl  = (const float*)d_in[3];
  const float* bl  = (const float*)d_in[4];
  const float* Wr  = (const float*)d_in[5];
  const float* br  = (const float*)d_in[6];
  const float* att = (const float*)d_in[7];
  const float* cb  = (const float*)d_in[8];
  const float* W1  = (const float*)d_in[9];
  const float* b1  = (const float*)d_in[10];
  const float* W2  = (const float*)d_in[11];
  const float* b2  = (const float*)d_in[12];
  float* h = (float*)d_out;                    // h state lives in d_out

  const int nE = in_sizes[1] / 2;
  const int nN = in_sizes[0] / TSTEPS;
  const int tot = nE + nN;

  char* ws = (char*)d_ws;
  _Float16* W1T = (_Float16*)ws;  ws += (size_t)256 * 256 * 2;
  _Float16* W2T = (_Float16*)ws;  ws += (size_t)128 * 256 * 2;
  unsigned* emaxKey = (unsigned*)ws; ws += (size_t)2 * nN * 4;
  float* den  = (float*)ws; ws += (size_t)2 * nN * 4;
  float* num  = (float*)ws; ws += (size_t)2 * nN * 4;
  float* eScr = (float*)ws; ws += (size_t)2 * tot * 4;
  float* uBuf = (float*)ws; ws += (size_t)nN * HIDDEN * 4;

  prep_weights<<<(65536 + 255) / 256, 256, 0, stream>>>(W1, W2, W1T, W2T);
  zero_h<<<(nN * HIDDEN + 255) / 256, 256, 0, stream>>>(h, nN * HIDDEN);

  const int nbE = (tot + 255) / 256;
  const int nbZ = (2 * nN + 255) / 256;
  const int nbG = (nN + 63) / 64;
  for (int t = 0; t < TSTEPS; ++t) {
    zero_seg<<<nbZ, 256, 0, stream>>>(emaxKey, den, num, 2 * nN);
    gat_edge_pass1<<<nbE, 256, 0, stream>>>(x, ei, Wl, Wr, bl, br, att,
                                            eScr, emaxKey, nE, nN, t);
    gat_edge_pass2<<<nbE, 256, 0, stream>>>(x, ei, eScr, emaxKey, den, num,
                                            nE, nN, t);
    gru_kernel<<<nbG, 128, 0, stream>>>(num, den, Wl, bl, cb, b1, b2,
                                        W1T, W2T, uBuf, h, nN);
  }
}
